// NetTransform_52166672777667
// MI455X (gfx1250) — matrix operand order repaired and checked by reading, not compiled, not run on hardware
//
#include <hip/hip_runtime.h>
#include <hip/hip_bf16.h>
#include <math.h>

typedef __attribute__((ext_vector_type(16))) _Float16 v16h;
typedef __attribute__((ext_vector_type(8)))  float    v8f;

#define WMMA_F16(a, b, c) \
  __builtin_amdgcn_wmma_f32_16x16x32_f16(false, (a), false, (b), (short)0, (c), false, false)

static constexpr int Bb   = 8;
static constexpr int T    = 256;
static constexpr int IDIM = 80;
static constexpr int HDIM = 512;
static constexpr int TNUM = 4;
static constexpr int BT   = Bb * T;          // 2048 (b,t) rows
static constexpr int BNT  = Bb * TNUM * T;   // 8192 (b,n,t) rows

// ---------------------------------------------------------------------------
// WMMA tile loaders (wave32; layouts per CDNA5 ISA §7.12.2)
// ---------------------------------------------------------------------------

// A-tile 16x32 f16 from an f32 matrix, hi/lo split. A[r][k] = src[r*ldr + k*ldk].
__device__ inline void load_a_f32_split(const float* __restrict__ src, int ldr, int ldk,
                                        int kbase, int kmax, int lane,
                                        v16h& hi, v16h& lo) {
  const int r = lane & 15, sel = lane >> 4;
#pragma unroll
  for (int v = 0; v < 8; ++v) {
    const int k0 = ((v >> 2) << 4) + (sel << 3) + ((v & 3) << 1);
#pragma unroll
    for (int p = 0; p < 2; ++p) {
      const int k = kbase + k0 + p;
      const float x = (k < kmax) ? src[r * ldr + k * ldk] : 0.0f;
      const _Float16 xh = (_Float16)x;
      hi[2 * v + p] = xh;
      lo[2 * v + p] = (_Float16)(x - (float)xh);
    }
  }
}

// B-tile 32x16 f16 from an f32 matrix, hi/lo split. B[k][c] = src[k*ldk + c*ldc].
__device__ inline void load_b_f32_split(const float* __restrict__ src, int ldk, int ldc,
                                        int kbase, int kmax, int lane,
                                        v16h& hi, v16h& lo, bool absval) {
  const int c = lane & 15, sel = lane >> 4;
#pragma unroll
  for (int v = 0; v < 8; ++v) {
#pragma unroll
    for (int p = 0; p < 2; ++p) {
      const int k = kbase + ((v >> 2) << 4) + (sel << 3) + ((v & 3) << 1) + p;
      float x = (k < kmax) ? src[k * ldk + c * ldc] : 0.0f;
      if (absval) x = fabsf(x);
      const _Float16 xh = (_Float16)x;
      hi[2 * v + p] = xh;
      lo[2 * v + p] = (_Float16)(x - (float)xh);
    }
  }
}

// A-tile 16x32 directly from an f16 matrix (row stride ldr, unit k stride).
__device__ inline void load_a_f16(const _Float16* __restrict__ src, int ldr,
                                  int kbase, int kmax, int lane, v16h& out, bool absval) {
  const int r = lane & 15, sel = lane >> 4;
#pragma unroll
  for (int v = 0; v < 8; ++v) {
    const int k0 = ((v >> 2) << 4) + (sel << 3) + ((v & 3) << 1);
#pragma unroll
    for (int p = 0; p < 2; ++p) {
      const int k = kbase + k0 + p;
      _Float16 x = (k < kmax) ? src[r * ldr + k] : (_Float16)0.0f;
      if (absval && x < (_Float16)0.0f) x = -x;
      out[2 * v + p] = x;
    }
  }
}

// B-tile 32x16 from an f16 matrix read transposed: B[k][c] = src[c*ldc + k].
__device__ inline void load_b_f16_t(const _Float16* __restrict__ src, int ldc,
                                    int kbase, int lane, v16h& out) {
  const int c = lane & 15, sel = lane >> 4;
#pragma unroll
  for (int v = 0; v < 8; ++v) {
#pragma unroll
    for (int p = 0; p < 2; ++p) {
      const int k = kbase + ((v >> 2) << 4) + (sel << 3) + ((v & 3) << 1) + p;
      out[2 * v + p] = src[c * ldc + k];
    }
  }
}

// ---------------------------------------------------------------------------
// K1: h = relu(X @ W_e + b_e), stored as f16 hi/lo pair for the next GEMM.
// Tiles: 128(M) x 32(N); one 16x16 tile per wave, 4 waves/block.
// ---------------------------------------------------------------------------
__global__ void k1_gemm_e(const float* __restrict__ X, const float* __restrict__ We,
                          const float* __restrict__ be,
                          _Float16* __restrict__ hhi, _Float16* __restrict__ hlo) {
  const int lane = threadIdx.x & 31;
  const int tile = blockIdx.x * 4 + (threadIdx.x >> 5);
  const int tn = tile & 31;
  const int tm = tile >> 5;
  const int r0 = tm * 16, c0 = tn * 16;
  v8f acc = {};
#pragma unroll
  for (int kb = 0; kb < IDIM; kb += 32) {
    v16h ahi, alo, bhi, blo;
    load_a_f32_split(X + r0 * IDIM, IDIM, 1, kb, IDIM, lane, ahi, alo);
    load_b_f32_split(We + c0, HDIM, 1, kb, IDIM, lane, bhi, blo, false);
    acc = WMMA_F16(ahi, bhi, acc);
    acc = WMMA_F16(ahi, blo, acc);
    acc = WMMA_F16(alo, bhi, acc);
  }
  const int col = c0 + (lane & 15);
  const int sel = lane >> 4;
  const float bias = be[col];
#pragma unroll
  for (int v = 0; v < 8; ++v) {
    const int row = r0 + v + 8 * sel;
    float h = acc[v] + bias;
    h = h > 0.0f ? h : 0.0f;
    const _Float16 hh = (_Float16)h;
    hhi[row * HDIM + col] = hh;
    hlo[row * HDIM + col] = (_Float16)(h - (float)hh);
  }
}

// ---------------------------------------------------------------------------
// K2: kernel[b,0,t,s] = sigmoid(sum_h h[b,t,h]*h[b,s,h]).  8 x (256x512)@(512x256).
// ---------------------------------------------------------------------------
__global__ void k2_kernel(const _Float16* __restrict__ hhi, const _Float16* __restrict__ hlo,
                          float* __restrict__ out) {
  const int lane = threadIdx.x & 31;
  const int tile = blockIdx.x * 4 + (threadIdx.x >> 5);   // 2048 tiles
  const int tj = tile & 15;
  const int ti = (tile >> 4) & 15;
  const int b  = tile >> 8;
  const int i0 = ti * 16, j0 = tj * 16;
  const _Float16* Ahi = hhi + (size_t)(b * T + i0) * HDIM;
  const _Float16* Alo = hlo + (size_t)(b * T + i0) * HDIM;
  const _Float16* Bhi = hhi + (size_t)(b * T + j0) * HDIM;
  const _Float16* Blo = hlo + (size_t)(b * T + j0) * HDIM;
  v8f acc = {};
  for (int kb = 0; kb < HDIM; kb += 32) {
    v16h ahi, alo, bhi, blo;
    load_a_f16(Ahi, HDIM, kb, HDIM, lane, ahi, false);
    load_a_f16(Alo, HDIM, kb, HDIM, lane, alo, false);
    load_b_f16_t(Bhi, HDIM, kb, lane, bhi);
    load_b_f16_t(Blo, HDIM, kb, lane, blo);
    acc = WMMA_F16(ahi, bhi, acc);
    acc = WMMA_F16(ahi, blo, acc);
    acc = WMMA_F16(alo, bhi, acc);
  }
  const int col = j0 + (lane & 15);
  const int sel = lane >> 4;
#pragma unroll
  for (int v = 0; v < 8; ++v) {
    const int row = i0 + v + 8 * sel;
    out[((size_t)(b * T + row)) * T + col] = 1.0f / (1.0f + expf(-acc[v]));
  }
}

// ---------------------------------------------------------------------------
// K3: pre_t = X @ W_t[n] + b_t[n]; out_hat = relu(pre_t)+x; m_t sign mask;
//     deterministic per-block loss partials. Tiles: n(4) x tm(128) x to(5).
// ---------------------------------------------------------------------------
__global__ void k3_gemm_t(const float* __restrict__ X, const float* __restrict__ Y4,
                          const float* __restrict__ Wt, const float* __restrict__ bt,
                          const int* __restrict__ ilens,
                          float* __restrict__ outhat, _Float16* __restrict__ m,
                          float* __restrict__ partials) {
  const int lane = threadIdx.x & 31;
  const int tile = blockIdx.x * 4 + (threadIdx.x >> 5);   // 2560 tiles
  const int to = tile % 5;
  const int tm = (tile / 5) & 127;
  const int n  = tile / (5 * 128);
  const int r0 = tm * 16, o0 = to * 16;
  const float* W = Wt + n * IDIM * IDIM;
  v8f acc = {};
#pragma unroll
  for (int kb = 0; kb < IDIM; kb += 32) {
    v16h ahi, alo, bhi, blo;
    load_a_f32_split(X + r0 * IDIM, IDIM, 1, kb, IDIM, lane, ahi, alo);
    load_b_f32_split(W + o0, IDIM, 1, kb, IDIM, lane, bhi, blo, false);
    acc = WMMA_F16(ahi, bhi, acc);
    acc = WMMA_F16(ahi, blo, acc);
    acc = WMMA_F16(alo, bhi, acc);
  }
  const int o   = o0 + (lane & 15);
  const int sel = lane >> 4;
  const float bias = bt[n * IDIM + o];
  float errsum = 0.0f;
#pragma unroll
  for (int v = 0; v < 8; ++v) {
    const int rg = r0 + v + 8 * sel;          // = b*T + t
    const int b = rg >> 8, t = rg & 255;
    const float wx  = acc[v];                 // (x @ W_t) without bias
    const float pre = wx + bias;
    const float rl  = pre > 0.0f ? pre : 0.0f;
    const float oh  = rl + X[rg * IDIM + o];
    outhat[((size_t)((b * TNUM + n) * T + t)) * IDIM + o] = oh;
    float mv = 0.0f;
    if (pre > 0.0f) mv = (wx > 0.0f) ? 1.0f : ((wx < 0.0f) ? -1.0f : 0.0f);
    m[((size_t)((b * TNUM + n) * T + t)) * IDIM + o] = (_Float16)mv;
    if (t < ilens[b]) {
      const float y = Y4[((size_t)((b * T + t) * TNUM + n)) * IDIM + o];
      const float d = oh - y;
      errsum += d * d;
    }
  }
  __shared__ float red[128];
  red[threadIdx.x] = errsum;
  __syncthreads();
  for (int s = 64; s > 0; s >>= 1) {
    if (threadIdx.x < s) red[threadIdx.x] += red[threadIdx.x + s];
    __syncthreads();
  }
  if (threadIdx.x == 0) partials[blockIdx.x] = red[0];
}

// ---------------------------------------------------------------------------
// K4: S1 = |m| @ |W_t|ᵀ, S2 = m @ W_tᵀ  →  P=(S1+S2)/2, N=(S1−S2)/2,
//     sign-selected |x|-weighted contribs. Tiles: 512 rowtiles x 5 d-tiles.
// ---------------------------------------------------------------------------
__global__ void k4_energy(const float* __restrict__ X, const float* __restrict__ Wt,
                          const _Float16* __restrict__ m,
                          float* __restrict__ numc, float* __restrict__ denc) {
  const int lane = threadIdx.x & 31;
  const int tile = blockIdx.x * 4 + (threadIdx.x >> 5);   // 2560 tiles
  const int td = tile % 5;
  const int tr = tile / 5;          // 0..511 over 8192 (b,n,t) rows
  const int r0 = tr * 16, d0 = td * 16;
  const int n  = (r0 >> 8) & 3;
  const float* W = Wt + n * IDIM * IDIM;
  v8f s1 = {}, s2 = {};
#pragma unroll
  for (int kb = 0; kb < IDIM; kb += 32) {
    v16h am, aa, bhi, blo, bahi, balo;
    load_a_f16(m + (size_t)r0 * IDIM, IDIM, kb, IDIM, lane, am, false);
    load_a_f16(m + (size_t)r0 * IDIM, IDIM, kb, IDIM, lane, aa, true);
    // B[k=o][c=d] = W_t[n][d0+c][o]
    load_b_f32_split(W + d0 * IDIM, 1, IDIM, kb, IDIM, lane, bhi, blo, false);
    load_b_f32_split(W + d0 * IDIM, 1, IDIM, kb, IDIM, lane, bahi, balo, true);
    s2 = WMMA_F16(am, bhi, s2);
    s2 = WMMA_F16(am, blo, s2);
    s1 = WMMA_F16(aa, bahi, s1);
    s1 = WMMA_F16(aa, balo, s1);
  }
  const int d   = d0 + (lane & 15);
  const int sel = lane >> 4;
#pragma unroll
  for (int v = 0; v < 8; ++v) {
    const int rg = r0 + v + 8 * sel;          // (b*4+n)*256 + t
    const int t  = rg & 255;
    const int b  = rg >> 10;
    const float P  = 0.5f * (s1[v] + s2[v]);
    const float Nn = 0.5f * (s1[v] - s2[v]);
    const float xv = X[((size_t)(b * T + t)) * IDIM + d];
    float nc = 0.0f, dc = 0.0f;
    if (xv > 0.0f)      { nc =  xv * Nn; dc =  xv * P;  }
    else if (xv < 0.0f) { nc = -xv * P;  dc = -xv * Nn; }
    numc[(size_t)rg * IDIM + d] = nc;
    denc[(size_t)rg * IDIM + d] = dc;
  }
}

// K5a: reduce over d, energy ratio, NaN->1, e = 1-(1-e)^2
__global__ void k5a_energy_raw(const float* __restrict__ numc, const float* __restrict__ denc,
                               float* __restrict__ ent) {
  const int i = blockIdx.x * blockDim.x + threadIdx.x;
  if (i >= BNT) return;
  float num = 0.0f, den = 0.0f;
  for (int d = 0; d < IDIM; ++d) {
    num += numc[(size_t)i * IDIM + d];
    den += denc[(size_t)i * IDIM + d];
  }
  float e = num / den;          // mean/mean: 1/80 cancels
  if (e != e) e = 1.0f;         // NaN -> 1
  const float om = 1.0f - e;
  ent[i] = 1.0f - om * om;
}

// K5b: global min/max over (B,4,T) for _minimaxn
__global__ void k5b_minmax(const float* __restrict__ ent, float* __restrict__ mm) {
  __shared__ float smin[1024], smax[1024];
  float mn = 3.0e38f, mx = -3.0e38f;
  for (int i = threadIdx.x; i < BNT; i += 1024) {
    const float v = ent[i];
    mn = fminf(mn, v); mx = fmaxf(mx, v);
  }
  smin[threadIdx.x] = mn; smax[threadIdx.x] = mx;
  __syncthreads();
  for (int s = 512; s > 0; s >>= 1) {
    if (threadIdx.x < s) {
      smin[threadIdx.x] = fminf(smin[threadIdx.x], smin[threadIdx.x + s]);
      smax[threadIdx.x] = fmaxf(smax[threadIdx.x], smax[threadIdx.x + s]);
    }
    __syncthreads();
  }
  if (threadIdx.x == 0) { mm[0] = smin[0]; mm[1] = smax[0]; }
}

// K6: loss = sum(partials) / (sum(ilens) * TNUM * IDIM)   (fixed-order, deterministic)
__global__ void k6_loss(const float* __restrict__ partials, int nparts,
                        const int* __restrict__ ilens, float* __restrict__ loss_out) {
  if (threadIdx.x == 0 && blockIdx.x == 0) {
    double s = 0.0;
    for (int i = 0; i < nparts; ++i) s += (double)partials[i];
    int nv = 0;
    for (int b = 0; b < Bb; ++b) nv += ilens[b];
    loss_out[0] = (float)(s / ((double)nv * TNUM * IDIM));
  }
}

// K7: _fb directional cumprods. JAX tril/triu both include the diagonal, so the
// pivot factor is (x[j]+1) and result[j][j] = x[j] exactly. One column per thread.
__global__ void k7_fb(const float* __restrict__ ent, const float* __restrict__ mm,
                      float* __restrict__ out) {
  __shared__ float xs[T];
  const int b = blockIdx.x;
  const int j = threadIdx.x;
  const float mn = mm[0], mx = mm[1];
  const float inv = 1.0f / (mx - mn);
  xs[j] = (ent[(b * TNUM + 0) * T + j] - mn) * inv;   // energy_t[b, 0, j] normalized
  __syncthreads();
  float* ob = out + (size_t)b * T * T;
  const float xj = xs[j];
  ob[(size_t)j * T + j] = xj;                         // 0.5*(x_f+x_b) on diag = x[j]
  float v = xj + 1.0f;
  for (int i = j + 1; i < T; ++i) { v *= xs[i]; ob[(size_t)i * T + j] = 0.5f * v; }
  float w = xj + 1.0f;
  for (int i = j - 1; i >= 0; --i) { w *= xs[i]; ob[(size_t)i * T + j] = 0.5f * w; }
}

// ---------------------------------------------------------------------------
extern "C" void kernel_launch(void* const* d_in, const int* in_sizes, int n_in,
                              void* d_out, int out_size, void* d_ws, size_t ws_size,
                              hipStream_t stream) {
  (void)in_sizes; (void)n_in; (void)out_size; (void)ws_size;
  const float* X   = (const float*)d_in[0];   // xs_pad_in  (8,256,80)
  const float* Y4  = (const float*)d_in[1];   // xs_pad_out (8,256,4,80)
  const float* We  = (const float*)d_in[2];   // W_e (80,512)
  const float* be  = (const float*)d_in[3];   // b_e (512,)
  const float* Wt  = (const float*)d_in[4];   // W_t (4,80,80)
  const float* bt  = (const float*)d_in[5];   // b_t (4,80)
  const int*   il  = (const int*)d_in[6];     // ilens (8,)
  // d_in[7] = ys_pad: unused by the forward pass

  float* out    = (float*)d_out;
  float* lossp  = out;                                    // [1]
  float* outhat = out + 1;                                // (8,4,256,80)
  float* kern   = outhat + (size_t)Bb * TNUM * T * IDIM;  // (8,1,256,256)
  float* ktar   = kern + (size_t)Bb * T * T;              // (8,1,256,256)

  char* ws = (char*)d_ws;
  _Float16* hhi  = (_Float16*)(ws + 0);           // 2048*512*2 = 2,097,152 B
  _Float16* hlo  = (_Float16*)(ws + 2097152);     // 2,097,152 B
  _Float16* mbuf = (_Float16*)(ws + 4194304);     // 8192*80*2 = 1,310,720 B
  float*    numc = (float*)(ws + 5505024);        // 8192*80*4 = 2,621,440 B
  float*    denc = (float*)(ws + 8126464);        // 2,621,440 B
  float*    entr = (float*)(ws + 10747904);       // 8192*4 = 32,768 B
  float*    mm   = (float*)(ws + 10780672);       // 2 floats
  float*    part = (float*)(ws + 10780680);       // 640 floats

  // Stage 1: h (split f16)                128x32 tiles, 4 waves/block
  k1_gemm_e<<<1024, 128, 0, stream>>>(X, We, be, hhi, hlo);
  // Stage 2: kernel = sigmoid(h hᵀ)       8x16x16 tiles
  k2_kernel<<<512, 128, 0, stream>>>(hhi, hlo, kern);
  // Stage 3: pre_t / out_hat / m / loss partials
  k3_gemm_t<<<640, 128, 0, stream>>>(X, Y4, Wt, bt, il, outhat, mbuf, part);
  // Stage 4: energy GEMMs (S1,S2) -> per-(b,n,t,d) contribs
  k4_energy<<<640, 128, 0, stream>>>(X, Wt, mbuf, numc, denc);
  // Stage 5: energy finalize + global min/max
  k5a_energy_raw<<<32, 256, 0, stream>>>(numc, denc, entr);
  k5b_minmax<<<1, 1024, 0, stream>>>(entr, mm);
  // Stage 6: loss scalar (deterministic serial combine)
  k6_loss<<<1, 64, 0, stream>>>(part, 640, il, lossp);
  // Stage 7: kernel_target_t via directional scans
  k7_fb<<<Bb, T, 0, stream>>>(entr, mm, ktar);
}
